// APPAP_5617817223943
// MI455X (gfx1250) — matrix operand order repaired and checked by reading, not compiled, not run on hardware
//
#include <hip/hip_runtime.h>
#include <hip/hip_bf16.h>
#include <cstdint>
#include <cstddef>

// ---------------------------------------------------------------------------
// MI455X (gfx1250) implementation:
//   ASPP (f16 WMMA implicit GEMM, async-to-LDS double-buffered weights)
//   -> q/k/v 1x1 convs (WMMA GEMM)
//   -> flash-attention fused softmax (WMMA QK^T and PV, no NxN materialized),
//      K blocks async-staged into double-buffered LDS shared by 4 waves.
// All matrix math uses v_wmma_f32_16x16x32_f16 (f32 accumulate).
// ---------------------------------------------------------------------------

typedef _Float16 f16;
typedef __attribute__((ext_vector_type(16))) _Float16 v16h;
typedef __attribute__((ext_vector_type(8)))  float    v8f;

#define EPSBN 1e-5f

union V16H { v16h v; uint4 u[2]; _Float16 h[16]; };

static __device__ __forceinline__ v8f wmma_f16(v16h a, v16h b, v8f c) {
#if defined(__gfx1250__)
  // (neg_a, A, neg_b, B, c_mod, C, reuse_a, reuse_b)
  return __builtin_amdgcn_wmma_f32_16x16x32_f16(false, a, false, b, (short)0, c,
                                                false, false);
#else
  (void)a; (void)b;
  return c;
#endif
}

// Async 16B copy global -> LDS (CDNA5 GLOBAL_LOAD_ASYNC_TO_LDS_B128,
// tracked by ASYNCcnt). Fallback: plain copy.
static __device__ __forceinline__ void stage_async16(const f16* g, f16* l) {
#if defined(__gfx1250__)
  unsigned ldsoff = (unsigned)(uintptr_t)l;  // low 32 bits = LDS byte offset
  asm volatile("global_load_async_to_lds_b128 %0, %1, off"
               :: "v"(ldsoff), "v"(g)
               : "memory");
#else
  *reinterpret_cast<uint4*>(l) = *reinterpret_cast<const uint4*>(g);
#endif
}

static __device__ __forceinline__ void wait_asynccnt0() {
#if defined(__gfx1250__)
#if __has_builtin(__builtin_amdgcn_s_wait_asynccnt)
  __builtin_amdgcn_s_wait_asynccnt(0);
#else
  asm volatile("s_wait_asynccnt 0x0" ::: "memory");
#endif
#endif
}

// ---------------------------------------------------------------------------
// Weight packing: OIHW f32 -> [O][tap*C] f16 (K = tap*C + c), BN scale folded.
// ---------------------------------------------------------------------------
__global__ void pack_w_kernel(const float* __restrict__ w,
                              const float* __restrict__ scale,
                              f16* __restrict__ wp, int O, int C, int taps) {
  int i = blockIdx.x * blockDim.x + threadIdx.x;
  int total = O * C * taps;
  if (i >= total) return;
  int o   = i / (C * taps);
  int rem = i - o * (C * taps);
  int t   = rem / C;
  int c   = rem - t * C;
  float s = (scale != nullptr) ? (scale[o] * rsqrtf(1.0f + EPSBN)) : 1.0f;
  wp[(size_t)o * (C * taps) + (size_t)t * C + c] =
      (f16)(w[((size_t)o * C + c) * taps + t] * s);
}

// ---------------------------------------------------------------------------
// Global average pool: p5[b][c] = mean_n x[b][c][n]
// ---------------------------------------------------------------------------
__global__ void meanpool_kernel(const float* __restrict__ x,
                                float* __restrict__ p5) {
  int bc = blockIdx.x;  // b*512 + c
  const float* src = x + (size_t)bc * 4096;
  float s = 0.f;
  for (int i = threadIdx.x; i < 4096; i += 256) s += src[i];
  __shared__ float red[256];
  red[threadIdx.x] = s;
  __syncthreads();
  for (int st = 128; st > 0; st >>= 1) {
    if ((int)threadIdx.x < st) red[threadIdx.x] += red[threadIdx.x + st];
    __syncthreads();
  }
  if (threadIdx.x == 0) p5[bc] = red[0] * (1.0f / 4096.0f);
}

// ---------------------------------------------------------------------------
// Branch 5 vector: b5[b][o] = relu((w5[o]·p5[b]) * inv + bias)
// ---------------------------------------------------------------------------
__global__ void b5_kernel(const float* __restrict__ w5,
                          const float* __restrict__ scale,
                          const float* __restrict__ bias,
                          const float* __restrict__ p5,
                          f16* __restrict__ b5h) {
  int b = blockIdx.x, o = threadIdx.x;
  float acc = 0.f;
  const float* wr = w5 + (size_t)o * 512;
  const float* pr = p5 + (size_t)b * 512;
  for (int c = 0; c < 512; ++c) acc += wr[c] * pr[c];
  float inv = scale[o] * rsqrtf(1.0f + EPSBN);
  float y = fmaxf(acc * inv + bias[o], 0.0f);
  b5h[b * 128 + o] = (f16)y;
}

// ---------------------------------------------------------------------------
// NCHW f32 -> "NHWC" f16 transpose via LDS tile (channels contiguous per pixel)
// ---------------------------------------------------------------------------
__global__ void nchw_to_nhwc_f16(const float* __restrict__ x,
                                 f16* __restrict__ xh) {
  __shared__ float tile[32][65];
  int b  = blockIdx.z;
  int c0 = blockIdx.y * 32, n0 = blockIdx.x * 64;
  int t  = threadIdx.x;
  int cl = t >> 6, nl = t & 63;
#pragma unroll
  for (int i = 0; i < 8; i++) {
    int c = cl + i * 4;
    tile[c][nl] = x[((size_t)(b * 512 + c0 + c)) * 4096 + n0 + nl];
  }
  __syncthreads();
  int c2 = t & 31, nb = t >> 5;
#pragma unroll
  for (int i = 0; i < 8; i++) {
    int n = nb + i * 8;
    xh[((size_t)(b * 4096 + n0 + n)) * 512 + c0 + c2] = (f16)tile[c2][n];
  }
}

// ---------------------------------------------------------------------------
// Broadcast pooled branch into feat channels [512..640)
// ---------------------------------------------------------------------------
__global__ void fill_b5_kernel(const f16* __restrict__ b5h,
                               f16* __restrict__ feat) {
  int i = blockIdx.x * 256 + threadIdx.x;
  int o = i & 127;
  int n = (i >> 7) & 4095;
  int b = i >> 19;
  feat[((size_t)(b * 4096 + n)) * 640 + 512 + o] = b5h[b * 128 + o];
}

// ---------------------------------------------------------------------------
// Implicit-GEMM conv (1x1 or dilated 3x3) with WMMA 16x16x32 f16.
//   A = packed weights [O][K], double-buffered in LDS via async-to-LDS
//   B = NHWC f16 input, register ping-pong across k-chunks
//   OT: o-tiles per wave, OSPLIT: waves splitting O (1 => waves split N)
//   NHWC out: out[(b*4096+n)*ldc + choff + o]
//   CHW  out: out[(b*ldc + choff + o)*4096 + n]   (ldc = total channels)
// ---------------------------------------------------------------------------
template <int OT, int OSPLIT, bool RELU, bool CHW_OUT>
__global__ __launch_bounds__(128) void conv_gemm_kernel(
    const f16* __restrict__ in, int Cin, int taps, int dil,
    const f16* __restrict__ wp, const float* __restrict__ bias,
    f16* __restrict__ out, int ldc, int choff) {
  constexpr int OROWS = OT * OSPLIT * 16;
  // rows padded to 40 halfs (80B = 20 banks) to stagger LDS banks
  __shared__ __align__(16) f16 wlds[2][OROWS][40];

  const int b    = blockIdx.y;
  const int tid  = threadIdx.x;
  const int wave = tid >> 5, lane = tid & 31;
  const int hi   = lane >> 4;
  const int obase = (wave % OSPLIT) * (OT * 16);
  constexpr int NPB = 16 * (4 / OSPLIT);
  const int nbase = blockIdx.x * NPB + (wave / OSPLIT) * 16;
  const int nl = nbase + (lane & 15);
  const int h = nl >> 6, w = nl & 63;
  const int Ktot  = taps * Cin;
  const int CperT = Cin >> 5;           // 32-k chunks per tap
  const int NKC   = taps * CperT;       // total k-chunks

  const v8f vzero = {0.f, 0.f, 0.f, 0.f, 0.f, 0.f, 0.f, 0.f};
  v8f acc[OT];
#pragma unroll
  for (int i = 0; i < OT; i++) acc[i] = vzero;

  // Issue async staging of one 32-k weight chunk into LDS buffer `buf`.
  auto issue_stage = [&](int kc, int buf) {
    const int t  = kc / CperT;
    const int cb = (kc - t * CperT) * 32;
    const int nv = OROWS * 4;  // 16B segments
    for (int idx = tid; idx < nv; idx += 128) {
      int row = idx >> 2, seg = idx & 3;
      stage_async16(wp + (size_t)row * Ktot + (size_t)t * Cin + cb + seg * 8,
                    &wlds[buf][row][seg * 8]);
    }
  };

  // Load B operand (16 contiguous K halves per lane) for k-chunk kc.
  auto load_b = [&](V16H& bm, int kc) {
    const int t  = kc / CperT;
    const int cb = (kc - t * CperT) * 32;
    const int dh = (taps == 9) ? ((t / 3) - 1) * dil : 0;
    const int dw = (taps == 9) ? ((t % 3) - 1) * dil : 0;
    const int hh = h + dh, ww = w + dw;
    const bool valid = (hh >= 0) && (hh < 64) && (ww >= 0) && (ww < 64);
    if (valid) {
      const uint4* p = reinterpret_cast<const uint4*>(
          in + ((size_t)(b * 4096 + hh * 64 + ww) * Cin + cb + hi * 16));
      bm.u[0] = p[0];
      bm.u[1] = p[1];
    } else {
#pragma unroll
      for (int z = 0; z < 16; z++) bm.h[z] = (f16)0;
    }
  };

  V16H bm[2];
  issue_stage(0, 0);
  load_b(bm[0], 0);
  wait_asynccnt0();
  __syncthreads();

  for (int kc = 0; kc < NKC; ++kc) {
    const int pb = kc & 1;
    if (kc + 1 < NKC) {
      issue_stage(kc + 1, pb ^ 1);   // async -> other LDS buffer
      load_b(bm[pb ^ 1], kc + 1);    // global -> regs, overlaps compute
    }
    const uint4* wlds4 =
        reinterpret_cast<const uint4*>(&wlds[pb][0][0]);  // 5 uint4 / row
#pragma unroll
    for (int tl = 0; tl < OT; tl++) {
      const int orow = obase + tl * 16 + (lane & 15);
      V16H am;
      am.u[0] = wlds4[orow * 5 + 2 * hi];       // K {k0..k0+7}
      am.u[1] = wlds4[orow * 5 + 2 * hi + 1];   // K {k0+8..k0+15}
      acc[tl] = wmma_f16(am.v, bm[pb].v, acc[tl]);
    }
    wait_asynccnt0();   // our async writes for next chunk are done
    __syncthreads();    // everyone done reading current / writing next
  }

  // --- epilogue: bias (+ReLU), f16 stores ---
#pragma unroll
  for (int tl = 0; tl < OT; tl++) {
    if constexpr (CHW_OUT) {
#pragma unroll
      for (int r = 0; r < 8; r++) {
        int o = obase + tl * 16 + hi * 8 + r;
        float v = acc[tl][r] + bias[o];
        if (RELU) v = fmaxf(v, 0.f);
        out[((size_t)(b * ldc + choff + o)) * 4096 + nl] = (f16)v;
      }
    } else {
#pragma unroll
      for (int r = 0; r < 8; r += 2) {
        int o = obase + tl * 16 + hi * 8 + r;
        float v0 = acc[tl][r] + bias[o];
        float v1 = acc[tl][r + 1] + bias[o + 1];
        if (RELU) { v0 = fmaxf(v0, 0.f); v1 = fmaxf(v1, 0.f); }
        union { f16 h2[2]; unsigned int u; } pk;
        pk.h2[0] = (f16)v0;
        pk.h2[1] = (f16)v1;
        *reinterpret_cast<unsigned int*>(
            out + ((size_t)(b * 4096 + nl)) * ldc + choff + o) = pk.u;
      }
    }
  }
}

// ---------------------------------------------------------------------------
// Flash attention: block = 4 waves sharing one 16-query tile.
//   Wave w owns V-channel group w (128 channels).
//   K blocks (64 keys x 128 ch) are async-staged into double-buffered LDS
//   shared by all waves (4x traffic reduction + full-block prefetch).
//   S = Q K^T from LDS, online softmax, O += P V^T (V streamed from global).
//   out = gamma * O / l + x
// ---------------------------------------------------------------------------
__global__ __launch_bounds__(128) void flash_attn_kernel(
    const f16* __restrict__ qb, const f16* __restrict__ kb,
    const f16* __restrict__ vb, const float* __restrict__ x,
    const float* __restrict__ gamma, float* __restrict__ out) {
  // K tile: 64 rows x 128 halfs, rows padded to 136 halfs (272B = 68 banks)
  __shared__ __align__(16) f16 klds[2][64][136];
  // per-wave P tile, rows padded to 72 halfs (144B = 36 banks)
  __shared__ __align__(16) f16 plds[4][16][72];

  const int qt = blockIdx.x, b = blockIdx.y;
  const int tid  = threadIdx.x;
  const int wave = tid >> 5, lane = tid & 31;
  const int cg   = wave;           // V-channel group of this wave
  const int hi = lane >> 4, lo = lane & 15;
  const int qbase = qt * 16;

  const v8f vzero = {0.f, 0.f, 0.f, 0.f, 0.f, 0.f, 0.f, 0.f};

  // Issue async staging of K block starting at key index mb.
  auto issue_stage_k = [&](int mb, int buf) {
    for (int idx = tid; idx < 512; idx += 128) {   // 64 rows x 8 segs of 16B
      int row = idx >> 3, seg = idx & 7;
      stage_async16(kb + ((size_t)(b * 4096 + mb + row)) * 128 + seg * 8,
                    &klds[buf][row][seg * 8]);
    }
  };

  // Preload Q as A-operands (4 chunks of K=32 over the 128 q-channels)
  V16H aq[4];
  {
    const f16* qrow = qb + ((size_t)(b * 4096 + qbase + lo)) * 128;
#pragma unroll
    for (int i = 0; i < 4; i++) {
      aq[i].u[0] = *reinterpret_cast<const uint4*>(qrow + i * 32 + hi * 16);
      aq[i].u[1] = *reinterpret_cast<const uint4*>(qrow + i * 32 + hi * 16 + 8);
    }
  }

  v8f acc[8];
#pragma unroll
  for (int i = 0; i < 8; i++) acc[i] = vzero;
  float mrow[8], lrow[8];
#pragma unroll
  for (int r = 0; r < 8; r++) { mrow[r] = -3.0e38f; lrow[r] = 0.f; }

  // V-block operand loader: 2 chunks of K(m)=32 for channel tile t
  auto load_vblk = [&](V16H* dst, int mb, int t) {
    const f16* vrow = vb +
                      ((size_t)(b * 512 + cg * 128 + t * 16 + lo)) * 4096 +
                      mb + hi * 16;
#pragma unroll
    for (int cc = 0; cc < 2; cc++) {
      const uint4* p = reinterpret_cast<const uint4*>(vrow + cc * 32);
      dst[cc].u[0] = p[0];
      dst[cc].u[1] = p[1];
    }
  };

  issue_stage_k(0, 0);
  wait_asynccnt0();
  __syncthreads();

  int buf = 0;
  for (int mb = 0; mb < 4096; mb += 64, buf ^= 1) {
    if (mb + 64 < 4096) {
      issue_stage_k(mb + 64, buf ^ 1);  // async prefetch next K block
      __builtin_prefetch(
          vb + ((size_t)(b * 512 + cg * 128 + lo)) * 4096 + mb + 64, 0, 1);
    }

    // ---- S = Q K^T : 4 m-subtiles x 4 k-chunks, K operands from LDS ----
    v8f s[4];
#pragma unroll
    for (int j = 0; j < 4; j++) {
      v8f sj = vzero;
#pragma unroll
      for (int i = 0; i < 4; i++) {
        V16H bk;
        const uint4* p = reinterpret_cast<const uint4*>(
            &klds[buf][j * 16 + lo][i * 32 + hi * 16]);
        bk.u[0] = p[0];
        bk.u[1] = p[1];
        sj = wmma_f16(aq[i].v, bk.v, sj);
      }
      s[j] = sj;
    }

    // ---- online softmax over this 64-key block ----
#pragma unroll
    for (int r = 0; r < 8; r++) {
      float mx = fmaxf(fmaxf(s[0][r], s[1][r]), fmaxf(s[2][r], s[3][r]));
#pragma unroll
      for (int off = 1; off < 16; off <<= 1)
        mx = fmaxf(mx, __shfl_xor(mx, off, 32));
      float mn = fmaxf(mrow[r], mx);
      float sc = __expf(mrow[r] - mn);
      float p0 = __expf(s[0][r] - mn), p1 = __expf(s[1][r] - mn);
      float p2 = __expf(s[2][r] - mn), p3 = __expf(s[3][r] - mn);
      float ps = p0 + p1 + p2 + p3;
#pragma unroll
      for (int off = 1; off < 16; off <<= 1) ps += __shfl_xor(ps, off, 32);
      lrow[r] = lrow[r] * sc + ps;
      mrow[r] = mn;
#pragma unroll
      for (int t = 0; t < 8; t++) acc[t][r] *= sc;
      const int row = r + 8 * hi;  // D-layout row
      plds[wave][row][0 * 16 + lo] = (f16)p0;
      plds[wave][row][1 * 16 + lo] = (f16)p1;
      plds[wave][row][2 * 16 + lo] = (f16)p2;
      plds[wave][row][3 * 16 + lo] = (f16)p3;
    }

    // ---- reload P as A-operands (2 chunks of K(m)=32) ----
    V16H ap[2];
#pragma unroll
    for (int cc = 0; cc < 2; cc++) {
      ap[cc].u[0] =
          *reinterpret_cast<const uint4*>(&plds[wave][lo][cc * 32 + hi * 16]);
      ap[cc].u[1] = *reinterpret_cast<const uint4*>(
          &plds[wave][lo][cc * 32 + hi * 16 + 8]);
    }

    // ---- O += P V^T over 8 channel tiles, ping-pong V operands ----
    V16H vbA[2], vbB[2];
    load_vblk(vbA, mb, 0);
#pragma unroll
    for (int t = 0; t < 8; t++) {
      V16H* cur = (t & 1) ? vbB : vbA;
      V16H* nxt = (t & 1) ? vbA : vbB;
      if (t < 7) load_vblk(nxt, mb, t + 1);
      acc[t] = wmma_f16(ap[0].v, cur[0].v, acc[t]);
      acc[t] = wmma_f16(ap[1].v, cur[1].v, acc[t]);
    }

    wait_asynccnt0();   // our portion of next K block staged
    __syncthreads();    // block done reading current K buffer
  }

  // ---- finalize: gamma * O/l + x ----
  const float g = gamma[0];
#pragma unroll
  for (int r = 0; r < 8; r++) {
    float invl = 1.0f / lrow[r];
    int n = qbase + r + 8 * hi;
#pragma unroll
    for (int t = 0; t < 8; t++) {
      int c = cg * 128 + t * 16 + lo;
      size_t idx = ((size_t)(b * 512 + c)) * 4096 + n;
      out[idx] = g * acc[t][r] * invl + x[idx];
    }
  }
}

// ---------------------------------------------------------------------------
// Host launcher
// ---------------------------------------------------------------------------
extern "C" void kernel_launch(void* const* d_in, const int* in_sizes, int n_in,
                              void* d_out, int out_size, void* d_ws,
                              size_t ws_size, hipStream_t stream) {
  (void)in_sizes; (void)n_in; (void)out_size; (void)ws_size;
  const float* x        = (const float*)d_in[0];
  const float* w_a1     = (const float*)d_in[1];
  const float* w_a2     = (const float*)d_in[2];
  const float* w_a3     = (const float*)d_in[3];
  const float* w_a4     = (const float*)d_in[4];
  const float* w_a5     = (const float*)d_in[5];
  const float* bn_scale = (const float*)d_in[6];
  const float* bn_bias  = (const float*)d_in[7];
  const float* w_q      = (const float*)d_in[8];
  const float* b_q      = (const float*)d_in[9];
  const float* w_k      = (const float*)d_in[10];
  const float* b_k      = (const float*)d_in[11];
  const float* w_v      = (const float*)d_in[12];
  const float* b_v      = (const float*)d_in[13];
  const float* gamma    = (const float*)d_in[14];
  float* out            = (float*)d_out;

  const int B = 4;
  char* ws = (char*)d_ws;
  // workspace layout (bytes)
  const size_t OFF_XH   = 0;                            // 4*4096*512*2
  const size_t OFF_FEAT = OFF_XH + 16777216ull;         // 4*4096*640*2
  const size_t OFF_Q    = OFF_FEAT + 20971520ull;       // 4*4096*128*2
  const size_t OFF_K    = OFF_Q + 4194304ull;
  const size_t OFF_V    = OFF_K + 4194304ull;           // 4*512*4096*2
  const size_t OFF_P5   = OFF_V + 16777216ull;          // 4*512*4
  const size_t OFF_B5H  = OFF_P5 + 8192ull;             // 4*128*2
  const size_t OFF_WA1  = OFF_B5H + 1024ull;            // 128*512*2
  const size_t OFF_WA2  = OFF_WA1 + 131072ull;          // 128*4608*2
  const size_t OFF_WA3  = OFF_WA2 + 1179648ull;
  const size_t OFF_WA4  = OFF_WA3 + 1179648ull;
  const size_t OFF_WQ   = OFF_WA4 + 1179648ull;         // 128*640*2
  const size_t OFF_WK   = OFF_WQ + 163840ull;
  const size_t OFF_WV   = OFF_WK + 163840ull;           // 512*512*2

  f16*   xh    = (f16*)(ws + OFF_XH);
  f16*   feat  = (f16*)(ws + OFF_FEAT);
  f16*   qbuf  = (f16*)(ws + OFF_Q);
  f16*   kbuf  = (f16*)(ws + OFF_K);
  f16*   vbuf  = (f16*)(ws + OFF_V);
  float* p5    = (float*)(ws + OFF_P5);
  f16*   b5h   = (f16*)(ws + OFF_B5H);
  f16*   wpa1  = (f16*)(ws + OFF_WA1);
  f16*   wpa2  = (f16*)(ws + OFF_WA2);
  f16*   wpa3  = (f16*)(ws + OFF_WA3);
  f16*   wpa4  = (f16*)(ws + OFF_WA4);
  f16*   wpq   = (f16*)(ws + OFF_WQ);
  f16*   wpk   = (f16*)(ws + OFF_WK);
  f16*   wpv   = (f16*)(ws + OFF_WV);

  auto pack = [&](const float* w, const float* sc, f16* wp, int O, int C,
                  int taps) {
    int total = O * C * taps;
    pack_w_kernel<<<(total + 255) / 256, 256, 0, stream>>>(w, sc, wp, O, C,
                                                           taps);
  };
  pack(w_a1, bn_scale + 0,   wpa1, 128, 512, 1);
  pack(w_a2, bn_scale + 128, wpa2, 128, 512, 9);
  pack(w_a3, bn_scale + 256, wpa3, 128, 512, 9);
  pack(w_a4, bn_scale + 384, wpa4, 128, 512, 9);
  pack(w_q,  nullptr,        wpq,  128, 640, 1);
  pack(w_k,  nullptr,        wpk,  128, 640, 1);
  pack(w_v,  nullptr,        wpv,  512, 512, 1);

  meanpool_kernel<<<B * 512, 256, 0, stream>>>(x, p5);
  b5_kernel<<<B, 128, 0, stream>>>(w_a5, bn_scale + 512, bn_bias + 512, p5,
                                   b5h);
  nchw_to_nhwc_f16<<<dim3(64, 16, B), 256, 0, stream>>>(x, xh);
  fill_b5_kernel<<<(B * 4096 * 128) / 256, 256, 0, stream>>>(b5h, feat);

  // ASPP conv branches -> feat channels [0..512)
  conv_gemm_kernel<8, 1, true, false><<<dim3(64, B), 128, 0, stream>>>(
      xh, 512, 1, 0, wpa1, bn_bias + 0, feat, 640, 0);
  conv_gemm_kernel<8, 1, true, false><<<dim3(64, B), 128, 0, stream>>>(
      xh, 512, 9, 2, wpa2, bn_bias + 128, feat, 640, 128);
  conv_gemm_kernel<8, 1, true, false><<<dim3(64, B), 128, 0, stream>>>(
      xh, 512, 9, 3, wpa3, bn_bias + 256, feat, 640, 256);
  conv_gemm_kernel<8, 1, true, false><<<dim3(64, B), 128, 0, stream>>>(
      xh, 512, 9, 6, wpa4, bn_bias + 384, feat, 640, 384);

  // q / k from feat (K=640) -> [b][n][128]
  conv_gemm_kernel<8, 1, false, false><<<dim3(64, B), 128, 0, stream>>>(
      feat, 640, 1, 0, wpq, b_q, qbuf, 128, 0);
  conv_gemm_kernel<8, 1, false, false><<<dim3(64, B), 128, 0, stream>>>(
      feat, 640, 1, 0, wpk, b_k, kbuf, 128, 0);

  // v from x (O=512) -> CHW layout [b][c][m]; two O=256 launches (LDS cap)
  conv_gemm_kernel<8, 2, false, true><<<dim3(128, B), 128, 0, stream>>>(
      xh, 512, 1, 0, wpv, b_v, vbuf, 512, 0);
  conv_gemm_kernel<8, 2, false, true><<<dim3(128, B), 128, 0, stream>>>(
      xh, 512, 1, 0, wpv + (size_t)256 * 512, b_v + 256, vbuf, 512, 256);

  // fused attention + residual
  flash_attn_kernel<<<dim3(256, B), 128, 0, stream>>>(qbuf, kbuf, vbuf, x,
                                                      gamma, out);
}
